// SelfAttention_9637906612676
// MI455X (gfx1250) — matrix operand order repaired and checked by reading, not compiled, not run on hardware
//
#include <hip/hip_runtime.h>
#include <hip/hip_bf16.h>

typedef __bf16 bf16_t;
typedef __attribute__((ext_vector_type(16))) __bf16 v16bf;
typedef __attribute__((ext_vector_type(8)))  __bf16 v8bf;
typedef __attribute__((ext_vector_type(8)))  float  v8f;

#define WMMA_BF16(a,b,c) __builtin_amdgcn_wmma_f32_16x16x32_bf16(false,(a),false,(b),(short)0,(c),false,false)

// ---- CDNA5 async global->LDS (ASYNCcnt-tracked), per CDNA5 ISA ----
__device__ __forceinline__ unsigned lds_off(const void* p) {
  return (unsigned)(size_t)p;              // low 32 bits of shared-aperture addr = LDS byte offset
}
__device__ __forceinline__ void async_b128(unsigned loff, const void* g) {
  asm volatile("global_load_async_to_lds_b128 %0, %1, off"
               :: "v"(loff), "v"((unsigned long long)(size_t)g) : "memory");
}
__device__ __forceinline__ void wait_async0() {
  asm volatile("s_wait_asynccnt 0x0" ::: "memory");
}

// ---- WMMA fragment loaders (layouts per CDNA5 ISA §7.12.2) ----
__device__ __forceinline__ v16bf load_frag_a(const bf16_t* p, int pitch) {
  int lane = threadIdx.x & 31;
  int row  = lane & 15;
  int koff = (lane >> 4) << 3;
  const bf16_t* r = p + row * pitch;
  union { v16bf v; v8bf h[2]; } u;
  u.h[0] = *(const v8bf*)(r + koff);
  u.h[1] = *(const v8bf*)(r + 16 + koff);
  return u.v;
}
__device__ __forceinline__ v16bf load_frag_b(const bf16_t* p, int pitch) {
  int lane = threadIdx.x & 31;
  int col  = lane & 15;
  int koff = (lane >> 4) << 3;
  const bf16_t* r = p + col * pitch;
  union { v16bf v; v8bf h[2]; } u;
  u.h[0] = *(const v8bf*)(r + koff);
  u.h[1] = *(const v8bf*)(r + 16 + koff);
  return u.v;
}

// ---- convert fp32 -> bf16 (optionally transposing a square matrix to [n][k]) ----
__global__ void __launch_bounds__(256) cvt_bf16_kernel(const float* __restrict__ src,
                                                       bf16_t* __restrict__ dst,
                                                       int n, int tdim) {
  int i = blockIdx.x * blockDim.x + threadIdx.x;
  if (i >= n) return;
  float v = src[i];
  if (tdim) {
    int r = i / tdim, c = i % tdim;
    dst[(size_t)c * tdim + r] = (bf16_t)v;
  } else {
    dst[i] = (bf16_t)v;
  }
}

// ---- shared GEMM body: [M=128] x [N=64] block tile, K-chunks of 32, async double-buffered ----
template <int OUTMODE>   // 0: bf16 head-split [B,H,N,64]; 1: fp32 row-major
__device__ __forceinline__ void gemm_body(const bf16_t* __restrict__ A,
                                          const bf16_t* __restrict__ Wt,
                                          bf16_t* __restrict__ outb,
                                          float* __restrict__ outf) {
  const int KDIM = 1024;
  __shared__ alignas(16) bf16_t At[2][128 * 40];
  __shared__ alignas(16) bf16_t Bt[2][64 * 40];
  int m0 = blockIdx.y * 128, n0 = blockIdx.x * 64;
  int tid = threadIdx.x, wid = tid >> 5, lane = tid & 31;
  int hi = lane >> 4, nl = lane & 15;

  // hoisted per-thread staging bases: thread covers rows r0, r0+32, ... ; seg kh constant
  int r0 = tid >> 2, kh = (tid & 3) * 8;
  const bf16_t* gA0 = A  + (size_t)(m0 + r0) * KDIM + kh;
  const bf16_t* gB0 = Wt + (size_t)(n0 + r0) * KDIM + kh;
  unsigned lA0 = lds_off(&At[0][r0 * 40 + kh]);
  unsigned lB0 = lds_off(&Bt[0][r0 * 40 + kh]);

  auto stage = [&](int kc, int bb) {
    const bf16_t* ga = gA0 + kc;
    const bf16_t* gb = gB0 + kc;
    unsigned la = lA0 + (unsigned)bb * (128u * 40u * 2u);
    unsigned lb = lB0 + (unsigned)bb * (64u * 40u * 2u);
#pragma unroll
    for (int it = 0; it < 4; ++it)                       // A: rows r0 + it*32 (128 rows)
      async_b128(la + it * (32 * 40 * 2), ga + (size_t)it * 32 * KDIM);
#pragma unroll
    for (int it = 0; it < 2; ++it)                       // B: rows r0 + it*32 (64 rows)
      async_b128(lb + it * (32 * 40 * 2), gb + (size_t)it * 32 * KDIM);
  };

  stage(0, 0);
  v8f acc[8] = {};
  int cur = 0;
  wait_async0();
  __syncthreads();
  for (int kc = 0; kc < KDIM; kc += 32) {
    if (kc + 32 < KDIM) stage(kc + 32, cur ^ 1);
    const bf16_t* Ab = At[cur];
    const bf16_t* Bb = Bt[cur];
    v16bf a0 = load_frag_a(Ab + (wid * 32) * 40, 40);
    v16bf a1 = load_frag_a(Ab + (wid * 32 + 16) * 40, 40);
#pragma unroll
    for (int nt = 0; nt < 4; ++nt) {
      v16bf b = load_frag_b(Bb + nt * 16 * 40, 40);
      acc[nt]     = WMMA_BF16(a0, b, acc[nt]);
      acc[4 + nt] = WMMA_BF16(a1, b, acc[4 + nt]);
    }
    wait_async0();
    __syncthreads();
    cur ^= 1;
  }

#pragma unroll
  for (int half = 0; half < 2; ++half)
#pragma unroll
    for (int nt = 0; nt < 4; ++nt)
#pragma unroll
      for (int r = 0; r < 8; ++r) {
        int gm = m0 + wid * 32 + half * 16 + r + 8 * hi;  // 0..4095
        int gc = n0 + nt * 16 + nl;                       // 0..1023
        float v = acc[half * 4 + nt][r];
        if (OUTMODE == 0) {
          int bb = gm >> 11, n = gm & 2047;
          int h = gc >> 6, dk = gc & 63;
          outb[(((size_t)bb * 16 + h) * 2048 + n) * 64 + dk] = (bf16_t)v;
        } else {
          outf[(size_t)gm * 1024 + gc] = v;
        }
      }
}

__global__ void __launch_bounds__(128) gemm_qkv_kernel(
    const bf16_t* __restrict__ Xb,
    const bf16_t* __restrict__ Wqt, const bf16_t* __restrict__ Wkt, const bf16_t* __restrict__ Wvt,
    bf16_t* __restrict__ Qo, bf16_t* __restrict__ Ko, bf16_t* __restrict__ Vo) {
  const bf16_t* W = (blockIdx.z == 0) ? Wqt : ((blockIdx.z == 1) ? Wkt : Wvt);
  bf16_t* Out     = (blockIdx.z == 0) ? Qo  : ((blockIdx.z == 1) ? Ko  : Vo);
  gemm_body<0>(Xb, W, Out, nullptr);
}

__global__ void __launch_bounds__(128) gemm_out_kernel(
    const bf16_t* __restrict__ A, const bf16_t* __restrict__ Wt, float* __restrict__ out) {
  gemm_body<1>(A, Wt, nullptr, out);
}

// ---- flash attention: per block one (b,h) x 64-query tile ----
__global__ void __launch_bounds__(128) attn_kernel(
    const bf16_t* __restrict__ Q, const bf16_t* __restrict__ K, const bf16_t* __restrict__ V,
    const float* __restrict__ bias,   // [H, N, N]
    bf16_t* __restrict__ O)           // [B, N, 1024]
{
  const int N = 2048, DK = 64;
  __shared__ alignas(16) bf16_t Qs[64 * 72];
  __shared__ alignas(16) bf16_t Ks[32 * 72];
  __shared__ alignas(16) bf16_t Vts[64 * 40];    // V transposed: [dk][key]
  __shared__ alignas(16) bf16_t Ps[4 * 16 * 40]; // per-wave P scratch, A layout
  int bh = blockIdx.y, b = bh >> 4, h = bh & 15;
  int q0 = blockIdx.x * 64;
  int tid = threadIdx.x, wid = tid >> 5, lane = tid & 31;
  int hi = lane >> 4, nl = lane & 15;
  const bf16_t* Qg = Q + ((size_t)bh * N + q0) * DK;
  const bf16_t* Kg = K + (size_t)bh * N * DK;
  const bf16_t* Vg = V + (size_t)bh * N * DK;

  // hoisted per-thread staging bases: row base tid>>3, seg (tid&7)*8 constant
  int sr = tid >> 3, seg = (tid & 7) * 8;
  const bf16_t* gQ0 = Qg + (size_t)sr * DK + seg;
  const bf16_t* gK0 = Kg + (size_t)sr * DK + seg;
  const bf16_t* gV0 = Vg + (size_t)sr * DK + seg;
  unsigned lQ0 = lds_off(&Qs[sr * 72 + seg]);
  unsigned lK0 = lds_off(&Ks[sr * 72 + seg]);

#pragma unroll
  for (int it = 0; it < 4; ++it)                 // Q tile 64x64: rows sr + it*16
    async_b128(lQ0 + it * (16 * 72 * 2), gQ0 + (size_t)it * 16 * DK);

  float mrow[8], lrow[8];
  v8f o[4] = {};
#pragma unroll
  for (int r = 0; r < 8; ++r) { mrow[r] = -1e30f; lrow[r] = 0.f; }

  for (int kt = 0; kt < N; kt += 32) {
#pragma unroll
    for (int it = 0; it < 2; ++it)               // K tile 32x64 [key][dk]: rows sr + it*16
      async_b128(lK0 + it * (16 * 72 * 2), gK0 + (size_t)(kt + it * 16) * DK);
#pragma unroll
    for (int it = 0; it < 2; ++it) {             // V tile transposed into Vts[dk][key]
      int ky = sr + it * 16;
      v8bf v = *(const v8bf*)(gV0 + (size_t)(kt + it * 16) * DK);
#pragma unroll
      for (int e = 0; e < 8; ++e) Vts[(seg + e) * 40 + ky] = v[e];
    }
    wait_async0();
    __syncthreads();

    // S = Q * K^T  (16 rows x 32 keys per wave), K-dim 64 in two 32-chunks
    v8f s0 = {}, s1 = {};
    v16bf aq0 = load_frag_a(Qs + wid * 16 * 72, 72);
    v16bf aq1 = load_frag_a(Qs + wid * 16 * 72 + 32, 72);
    v16bf bk;
    bk = load_frag_b(Ks, 72);                s0 = WMMA_BF16(aq0, bk, s0);
    bk = load_frag_b(Ks + 32, 72);           s0 = WMMA_BF16(aq1, bk, s0);
    bk = load_frag_b(Ks + 16 * 72, 72);      s1 = WMMA_BF16(aq0, bk, s1);
    bk = load_frag_b(Ks + 16 * 72 + 32, 72); s1 = WMMA_BF16(aq1, bk, s1);

    bf16_t* pl = Ps + wid * 16 * 40;
#pragma unroll
    for (int r = 0; r < 8; ++r) {
      int qg = q0 + wid * 16 + r + 8 * hi;
      const float* bp = bias + ((size_t)h * N + qg) * N + kt;
      float v0 = s0[r] + bp[nl];
      float v1 = s1[r] + bp[16 + nl];
      float mx = fmaxf(v0, v1);
#pragma unroll
      for (int d = 8; d >= 1; d >>= 1) mx = fmaxf(mx, __shfl_xor(mx, d, 32));
      float mnew  = fmaxf(mrow[r], mx);
      float scale = __expf(mrow[r] - mnew);
      float p0 = __expf(v0 - mnew);
      float p1 = __expf(v1 - mnew);
      float sum = p0 + p1;
#pragma unroll
      for (int d = 8; d >= 1; d >>= 1) sum += __shfl_xor(sum, d, 32);
      lrow[r] = lrow[r] * scale + sum;
      mrow[r] = mnew;
#pragma unroll
      for (int nt = 0; nt < 4; ++nt) o[nt][r] *= scale;
      pl[(r + 8 * hi) * 40 + nl]      = (bf16_t)p0;
      pl[(r + 8 * hi) * 40 + 16 + nl] = (bf16_t)p1;
    }
    asm volatile("s_wait_dscnt 0x0" ::: "memory");   // own P stores -> own ds_reads

    v16bf ap = load_frag_a(pl, 40);                  // O += P * V
#pragma unroll
    for (int nt = 0; nt < 4; ++nt) {
      v16bf bv = load_frag_b(Vts + nt * 16 * 40, 40);
      o[nt] = WMMA_BF16(ap, bv, o[nt]);
    }
    __syncthreads();                                 // all waves done with Ks/Vts
  }

#pragma unroll
  for (int nt = 0; nt < 4; ++nt)
#pragma unroll
    for (int r = 0; r < 8; ++r) {
      int qg = q0 + wid * 16 + r + 8 * hi;
      int col = h * 64 + nt * 16 + nl;
      O[((size_t)b * N + qg) * 1024 + col] = (bf16_t)(o[nt][r] / lrow[r]);
    }
}

extern "C" void kernel_launch(void* const* d_in, const int* in_sizes, int n_in,
                              void* d_out, int out_size, void* d_ws, size_t ws_size,
                              hipStream_t stream) {
  const float* x  = (const float*)d_in[0];
  const float* pb = (const float*)d_in[1];
  const float* Wq = (const float*)d_in[2];
  const float* Wk = (const float*)d_in[3];
  const float* Wv = (const float*)d_in[4];
  const float* Wo = (const float*)d_in[5];

  const size_t BN = 4096, D = 1024;
  char* ws = (char*)d_ws;
  bf16_t* xb  = (bf16_t*)ws; ws += BN * D * sizeof(bf16_t);
  bf16_t* wqt = (bf16_t*)ws; ws += D * D * sizeof(bf16_t);
  bf16_t* wkt = (bf16_t*)ws; ws += D * D * sizeof(bf16_t);
  bf16_t* wvt = (bf16_t*)ws; ws += D * D * sizeof(bf16_t);
  bf16_t* wot = (bf16_t*)ws; ws += D * D * sizeof(bf16_t);
  bf16_t* Qh  = (bf16_t*)ws; ws += BN * D * sizeof(bf16_t);
  bf16_t* Kh  = (bf16_t*)ws; ws += BN * D * sizeof(bf16_t);
  bf16_t* Vh  = (bf16_t*)ws; ws += BN * D * sizeof(bf16_t);
  bf16_t* Ob  = (bf16_t*)ws; ws += BN * D * sizeof(bf16_t);

  cvt_bf16_kernel<<<(int)((BN * D + 255) / 256), 256, 0, stream>>>(x,  xb,  (int)(BN * D), 0);
  cvt_bf16_kernel<<<(int)((D * D + 255) / 256), 256, 0, stream>>>(Wq, wqt, (int)(D * D), (int)D);
  cvt_bf16_kernel<<<(int)((D * D + 255) / 256), 256, 0, stream>>>(Wk, wkt, (int)(D * D), (int)D);
  cvt_bf16_kernel<<<(int)((D * D + 255) / 256), 256, 0, stream>>>(Wv, wvt, (int)(D * D), (int)D);
  cvt_bf16_kernel<<<(int)((D * D + 255) / 256), 256, 0, stream>>>(Wo, wot, (int)(D * D), (int)D);

  gemm_qkv_kernel<<<dim3(16, 32, 3), 128, 0, stream>>>(xb, wqt, wkt, wvt, Qh, Kh, Vh);
  attn_kernel<<<dim3(32, 32), 128, 0, stream>>>(Qh, Kh, Vh, pb, Ob);
  gemm_out_kernel<<<dim3(16, 32), 128, 0, stream>>>(Ob, wot, (float*)d_out);
}
